// RNNStateEncoder_87308095193561
// MI455X (gfx1250) — matrix operand order repaired and checked by reading, not compiled, not run on hardware
//
#include <hip/hip_runtime.h>
#include <hip/hip_bf16.h>
#include <math.h>

// ---------------------------------------------------------------------------
// Types for CDNA5 WMMA (wave32): v_wmma_f32_16x16x32_bf16
// ---------------------------------------------------------------------------
typedef __attribute__((ext_vector_type(16))) __bf16 bf16x16;
typedef __attribute__((ext_vector_type(8)))  float  f32x8;

#define TT 256   // timesteps
#define NE 128   // envs
#define II 1024  // input dim
#define HH 1024  // hidden dim
#define G3 3072  // 3*H

#define SBS 72    // padded LDS stride (elems) for B tiles  (144B = 9x16B)
#define SAS 1032  // padded LDS stride (elems) for A rows   (2064B = 129x16B)

union FragU { bf16x16 v; uint4 q[2]; };

// ---------------------------------------------------------------------------
// CDNA5 async global->LDS copy (ASYNCcnt) + waits
// ---------------------------------------------------------------------------
__device__ __forceinline__ void async_copy_b128(unsigned lds_byte_off, const void* gaddr) {
    unsigned long long ga = (unsigned long long)(uintptr_t)gaddr;
    asm volatile("global_load_async_to_lds_b128 %0, %1, off"
                 :: "v"(lds_byte_off), "v"(ga) : "memory");
}
__device__ __forceinline__ void wait_async_le(int n) {
    if (n == 0) asm volatile("s_wait_asynccnt 0x0" ::: "memory");
    else        asm volatile("s_wait_asynccnt 0x4" ::: "memory");
}

// A fragment: 16x32 bf16 tile, rows M=0..15 from row-major src (stride lda elems).
// lanes 0-15: M=lane, K=0..7 / 16..23; lanes 16-31: M=lane-16, K=8..15 / 24..31.
__device__ __forceinline__ bf16x16 load_a_frag(const __bf16* src, int lda, int lane) {
    const int half = lane & 15;
    const int hi   = lane >> 4;
    const __bf16* p0 = src + half * lda + hi * 8;
    FragU f;
    f.q[0] = *(const uint4*)(p0);
    f.q[1] = *(const uint4*)(p0 + 16);
    return f.v;
}

// B fragment: 32x16 tile, B[k][n] = W[nRow+n][k0+k] (row-major W, stride ldw).
// per lane: W row (nRow + lane&15), K = k0 + (lane>>4)*8 + 0..7 / 16..23.
__device__ __forceinline__ bf16x16 load_b_frag(const __bf16* W, int ldw,
                                               int nRow, int k0, int lane) {
    const int half = lane & 15;
    const int hi   = lane >> 4;
    const __bf16* p = W + (size_t)(nRow + half) * ldw + k0 + hi * 8;
    FragU f;
    f.q[0] = *(const uint4*)(p);
    f.q[1] = *(const uint4*)(p + 16);
    return f.v;
}

__device__ __forceinline__ f32x8 wmma_bf16(bf16x16 a, bf16x16 b, f32x8 c) {
    return __builtin_amdgcn_wmma_f32_16x16x32_bf16(
        false, a, false, b, (short)0, c, false, false);
}

// ---------------------------------------------------------------------------
// fp32 -> bf16 conversion (4 elems/thread, n multiple of 4)
// ---------------------------------------------------------------------------
__global__ void cvt_f32_bf16(const float* __restrict__ in, __bf16* __restrict__ out,
                             long n) {
    long i = ((long)blockIdx.x * blockDim.x + threadIdx.x) * 4;
    if (i + 3 < n) {
        float4 v = *(const float4*)(in + i);
        union { __bf16 b[4]; uint2 u; } o;
        o.b[0] = (__bf16)v.x; o.b[1] = (__bf16)v.y;
        o.b[2] = (__bf16)v.z; o.b[3] = (__bf16)v.w;
        *(uint2*)(out + i) = o.u;
    }
}

__global__ void zero_u32(unsigned* p) { *p = 0u; }

// ---------------------------------------------------------------------------
// gi = x_bf16 @ W_ih^T + b_ih : M=32768, N=3072, K=1024, fp32 out.
// Grid (256, 24) x 256 thr. B tile (128 cols x 64 K) staged in LDS via
// async global->LDS copies, double-buffered; shared by all 8 waves.
// Inner loop software-pipelined: B frag j+1 loads while WMMA j executes.
// ---------------------------------------------------------------------------
__global__ __launch_bounds__(256)
void gi_gemm(const __bf16* __restrict__ Xb, const __bf16* __restrict__ Wb,
             const float* __restrict__ bias, float* __restrict__ gi) {
    __shared__ __bf16 sB[2][128 * SBS];   // 2 x 18 KB

    const int tid   = threadIdx.x;
    const int lane  = tid & 31;
    const int wave  = tid >> 5;
    const int mBase = blockIdx.x * 128 + wave * 16;
    const int nBase = blockIdx.y * 128;

    // stage one 128x64 B chunk (kb) into sB[buf]: 4 async b128 ops/thread
    auto stage = [&](int kb, int buf) {
#pragma unroll
        for (int r4 = 0; r4 < 4; ++r4) {
            int idx = r4 * 256 + tid;           // 0..1023
            int row = idx >> 3;                  // 0..127
            int v   = idx & 7;                   // 16B vector within row
            const __bf16* g = Wb + (size_t)(nBase + row) * II + kb + v * 8;
            unsigned lo = (unsigned)(uintptr_t)&sB[buf][row * SBS + v * 8];
            async_copy_b128(lo, g);
        }
    };

    f32x8 acc[8];
#pragma unroll
    for (int n = 0; n < 8; ++n) acc[n] = (f32x8)0.0f;

    const __bf16* aRow = Xb + (size_t)mBase * II;

    stage(0, 0);
    for (int c = 0; c < 16; ++c) {               // 16 chunks of K=64
        const int kb = c * 64;
        if (c < 15) stage(kb + 64, (c + 1) & 1);

        // A loads issued before the async wait: global latency overlaps it
        bf16x16 a0 = load_a_frag(aRow + kb,      II, lane);
        bf16x16 a1 = load_a_frag(aRow + kb + 32, II, lane);
        __builtin_prefetch(aRow + kb + 128, 0, 0);

        wait_async_le(c < 15 ? 4 : 0);           // chunk c arrived (newest 4 fly)
        __syncthreads();

        const __bf16* bTile = &sB[c & 1][0];
        // 16 tiles flat: j = (ks>>5)*8 + n ; pipeline B frags one tile ahead
        bf16x16 bcur = load_b_frag(bTile, SBS, 0, 0, lane);
#pragma unroll
        for (int j = 0; j < 16; ++j) {
            bf16x16 bnxt = bcur;
            if (j + 1 < 16)
                bnxt = load_b_frag(bTile, SBS, ((j + 1) & 7) * 16,
                                   ((j + 1) >> 3) * 32, lane);
            acc[j & 7] = wmma_bf16((j >> 3) ? a1 : a0, bcur, acc[j & 7]);
            bcur = bnxt;
        }
        __syncthreads();                          // safe to overwrite sB[c&1]
    }

    const int half = lane & 15;
    const int hi   = lane >> 4;
#pragma unroll
    for (int n = 0; n < 8; ++n) {
        int col = nBase + n * 16 + half;
        float bv = bias[col];
#pragma unroll
        for (int r = 0; r < 8; ++r) {
            int row = mBase + hi * 8 + r;
            gi[(size_t)row * G3 + col] = acc[n][r] + bv;
        }
    }
}

// ---------------------------------------------------------------------------
// Persistent GRU scan: one launch, 64 WGs (8 env-groups x 8 col-groups),
// device-wide spin barrier between timesteps. Each step:
//   - stage masked h rows (16 x 1024) as bf16 into padded LDS
//   - 8 waves x (3 gates x K=1024) WMMA chains, A from LDS, B from L2,
//     software-pipelined one K-step ahead
//   - fused sigmoid/tanh gates, h_new -> out slab (read by step t+1)
// ---------------------------------------------------------------------------
__global__ __launch_bounds__(256)
void gru_scan(const float*  __restrict__ h0,
              const float*  __restrict__ masks,  // T*N
              const float*  __restrict__ gi,     // T*N x 3072
              const __bf16* __restrict__ Whb,    // 3072 x 1024 bf16
              const float*  __restrict__ bhh,    // 3072
              float* __restrict__ out,           // T*N x 1024
              unsigned* __restrict__ ctr) {
    __shared__ __bf16 sA[16 * SAS];              // ~33 KB, padded vs bank conflicts

    const int tid   = threadIdx.x;
    const int lane  = tid & 31;
    const int wave  = tid >> 5;
    const int eg    = blockIdx.x & 7;            // env group
    const int cg    = blockIdx.x >> 3;           // col group
    const int mBase = eg * 16;
    const int nBase = cg * 128 + wave * 16;

    const int half = lane & 15;
    const int hi   = lane >> 4;
    const int col  = nBase + half;
    const float br_ = bhh[col];
    const float bz_ = bhh[HH + col];
    const float bn_ = bhh[2 * HH + col];

    for (int t = 0; t < TT; ++t) {
        const float* hp     = (t == 0) ? h0 : out + (size_t)(t - 1) * NE * HH;
        const float* mask_t = masks + (size_t)t * NE;
        const float* gi_t   = gi + (size_t)t * NE * G3;
        float*       hout   = out + (size_t)t * NE * HH;

        // ---- stage masked h (16 rows x 1024) -> bf16 LDS, 4 elems/thread x16
#pragma unroll
        for (int it = 0; it < 16; ++it) {
            int i   = (it * 256 + tid) * 4;      // < 16384
            int row = i >> 10;
            int cc  = i & 1023;
            float m = mask_t[mBase + row];
            float4 v = *(const float4*)(hp + (size_t)(mBase + row) * HH + cc);
            union { __bf16 b[4]; uint2 u; } o;
            o.b[0] = (__bf16)(v.x * m); o.b[1] = (__bf16)(v.y * m);
            o.b[2] = (__bf16)(v.z * m); o.b[3] = (__bf16)(v.w * m);
            *(uint2*)&sA[row * SAS + cc] = o.u;
        }
        __syncthreads();

        // ---- 3 WMMA chains sharing A (from LDS), pipelined one K-step ahead
        f32x8 cr = (f32x8)0.0f, cz = (f32x8)0.0f, cn = (f32x8)0.0f;
        bf16x16 a  = load_a_frag(&sA[0], SAS, lane);
        bf16x16 br = load_b_frag(Whb, HH, nBase,          0, lane);
        bf16x16 bz = load_b_frag(Whb, HH, HH + nBase,     0, lane);
        bf16x16 bn = load_b_frag(Whb, HH, 2 * HH + nBase, 0, lane);
        for (int k0 = 0; k0 < HH - 32; k0 += 32) {
            const int k1 = k0 + 32;
            bf16x16 a2  = load_a_frag(&sA[0] + k1, SAS, lane);
            bf16x16 br2 = load_b_frag(Whb, HH, nBase,          k1, lane);
            bf16x16 bz2 = load_b_frag(Whb, HH, HH + nBase,     k1, lane);
            bf16x16 bn2 = load_b_frag(Whb, HH, 2 * HH + nBase, k1, lane);
            cr = wmma_bf16(a, br, cr);
            cz = wmma_bf16(a, bz, cz);
            cn = wmma_bf16(a, bn, cn);
            a = a2; br = br2; bz = bz2; bn = bn2;
        }
        cr = wmma_bf16(a, br, cr);
        cz = wmma_bf16(a, bz, cz);
        cn = wmma_bf16(a, bn, cn);

        // ---- gates + blend
#pragma unroll
        for (int r = 0; r < 8; ++r) {
            int row = mBase + hi * 8 + r;
            const float* gir = gi_t + (size_t)row * G3;
            float xr = gir[col]      + cr[r] + br_;
            float xz = gir[HH + col] + cz[r] + bz_;
            float rg = 1.0f / (1.0f + __expf(-xr));
            float zg = 1.0f / (1.0f + __expf(-xz));
            float ng = tanhf(gir[2 * HH + col] + rg * (cn[r] + bn_));
            float hm = hp[(size_t)row * HH + col] * mask_t[row];
            hout[(size_t)row * HH + col] = (1.0f - zg) * ng + zg * hm;
        }

        // ---- device-wide barrier (monotonic counter; 64 WGs all resident)
        __threadfence();
        __syncthreads();
        if (tid == 0) {
            __hip_atomic_fetch_add(ctr, 1u, __ATOMIC_ACQ_REL, __HIP_MEMORY_SCOPE_AGENT);
            unsigned target = 64u * (unsigned)(t + 1);
            while (__hip_atomic_load(ctr, __ATOMIC_ACQUIRE, __HIP_MEMORY_SCOPE_AGENT)
                   < target)
                __builtin_amdgcn_s_sleep(2);
        }
        __syncthreads();
    }
}

// ---------------------------------------------------------------------------
// Launch
// ---------------------------------------------------------------------------
extern "C" void kernel_launch(void* const* d_in, const int* in_sizes, int n_in,
                              void* d_out, int out_size, void* d_ws, size_t ws_size,
                              hipStream_t stream) {
    const float* x   = (const float*)d_in[0];  // 32768 x 1024
    const float* h0  = (const float*)d_in[1];  // 128 x 1 x 1024
    const float* mk  = (const float*)d_in[2];  // 32768
    const float* Wih = (const float*)d_in[3];  // 3072 x 1024
    const float* Whh = (const float*)d_in[4];  // 3072 x 1024
    const float* bih = (const float*)d_in[5];  // 3072
    const float* bhh = (const float*)d_in[6];  // 3072
    float* out = (float*)d_out;                // outs ++ h_final

    // Workspace layout (16B aligned slabs)
    char* ws = (char*)d_ws;
    unsigned* ctr   = (unsigned*)ws;  ws += 16;
    __bf16* Wih_b = (__bf16*)ws;      ws += (size_t)G3 * II * 2;       // 6 MB
    __bf16* Whh_b = (__bf16*)ws;      ws += (size_t)G3 * HH * 2;       // 6 MB
    __bf16* X_b   = (__bf16*)ws;      ws += (size_t)TT * NE * II * 2;  // 64 MB
    float*  gi    = (float*)ws;                                        // 384 MB

    zero_u32<<<1, 1, 0, stream>>>(ctr);
    cvt_f32_bf16<<<(G3 * II) / 1024, 256, 0, stream>>>(Wih, Wih_b, (long)G3 * II);
    cvt_f32_bf16<<<(G3 * HH) / 1024, 256, 0, stream>>>(Whh, Whh_b, (long)G3 * HH);
    cvt_f32_bf16<<<((size_t)TT * NE * II) / 1024, 256, 0, stream>>>(
        x, X_b, (long)TT * NE * II);

    gi_gemm<<<dim3(256, 24), 256, 0, stream>>>(X_b, Wih_b, bih, gi);

    gru_scan<<<64, 256, 0, stream>>>(h0, mk, gi, Whh_b, bhh, out, ctr);

    // h_final = outs[T-1]
    hipMemcpyAsync(out + (size_t)TT * NE * HH,
                   out + (size_t)(TT - 1) * NE * HH,
                   (size_t)NE * HH * sizeof(float),
                   hipMemcpyDeviceToDevice, stream);
}
